// MultiHeadAttentionBlock_9062380995221
// MI455X (gfx1250) — matrix operand order repaired and checked by reading, not compiled, not run on hardware
//
#include <hip/hip_runtime.h>

// MHA block for MI455X (gfx1250): bf16 WMMA (16x16x32), fp32 accum.
// B=2, S=2048, D=1024, H=16, DK=64. Softmax over HEADS (faithful to ref quirk):
// masked (k>q) positions get attn = 1/16 per head.

#define B_  2
#define S_  2048
#define D_  1024
#define H_  16
#define DK_ 64

typedef __attribute__((ext_vector_type(16))) __bf16 v16bf;
typedef __attribute__((ext_vector_type(8)))  __bf16 bf16x8;
typedef __attribute__((ext_vector_type(8)))  float  v8f;
typedef __attribute__((ext_vector_type(4)))  float  f32x4;

union V16U { v16bf v; bf16x8 h[2]; };

__device__ __forceinline__ v8f zero8() {
  v8f z = {0.f, 0.f, 0.f, 0.f, 0.f, 0.f, 0.f, 0.f};
  return z;
}

__device__ __forceinline__ v8f wmma_bf16(v16bf a, v16bf b, v8f c) {
  // D = A(16x32 bf16) * B(32x16 bf16) + C(16x16 f32)
  return __builtin_amdgcn_wmma_f32_16x16x32_bf16(
      /*neg_a=*/false, a, /*neg_b=*/false, b,
      /*c_mod=*/(short)0, c, /*reuse_a=*/false, /*reuse_b=*/false);
}

// Build a 16-element bf16 fragment from two 8-element (16B) chunks.
__device__ __forceinline__ v16bf frag2(const __bf16* p0, const __bf16* p1) {
  V16U u;
  u.h[0] = *(const bf16x8*)p0;
  u.h[1] = *(const bf16x8*)p1;
  return u.v;
}

// LDS byte offset of a generic pointer to __shared__ memory.
__device__ __forceinline__ unsigned lds_off(const void* p) {
  return (unsigned)(unsigned long long)(
      (__attribute__((address_space(3))) const void*)p);
}

// CDNA5 async global->LDS copy (16 bytes per lane), tracked with ASYNCcnt.
__device__ __forceinline__ void async_copy_b128(unsigned lds_addr,
                                                const void* gptr) {
  asm volatile("global_load_async_to_lds_b128 %0, %1, off"
               :
               : "v"(lds_addr), "v"((unsigned long long)gptr)
               : "memory");
}

__device__ __forceinline__ void wait_async0() {
#if __has_builtin(__builtin_amdgcn_s_wait_asynccnt)
  __builtin_amdgcn_s_wait_asynccnt(0);
#else
  asm volatile("s_wait_asynccnt 0x0" ::: "memory");
#endif
}

// Stage 32 fp32 -> 32 bf16 into LDS (one half-row of a 128x64 tile).
__device__ __forceinline__ void stage_cvt32(const float* __restrict__ p,
                                            __bf16* dst) {
  f32x4 f[8];
#pragma unroll
  for (int i = 0; i < 8; i++) f[i] = ((const f32x4*)p)[i];
#pragma unroll
  for (int j = 0; j < 4; j++) {
    bf16x8 o;
#pragma unroll
    for (int i = 0; i < 4; i++) {
      o[i]     = (__bf16)f[2 * j][i];
      o[i + 4] = (__bf16)f[2 * j + 1][i];
    }
    *(bf16x8*)(dst + 8 * j) = o;
  }
}

// ---------------------------------------------------------------------------
// Projection GEMM: C[M=4096, N=1024] = X @ W^T + bias, output bf16.
// vtrans=0: store as [b,h,s,dk]; vtrans=1: store V transposed [b,h,dk,s].
// 256 threads = 8 waves; tile 128x128; per-wave 64Mx32N; K-step 64.
// ---------------------------------------------------------------------------
__global__ __launch_bounds__(256)
void proj_kernel(const float* __restrict__ X, const float* __restrict__ W,
                 const float* __restrict__ bias, __bf16* __restrict__ Out,
                 int vtrans) {
  __shared__ __bf16 sA[128][64];
  __shared__ __bf16 sB[128][64];

  const int tid  = threadIdx.x;
  const int lane = tid & 31;
  const int wave = tid >> 5;
  const int m0 = blockIdx.y * 128;
  const int n0 = blockIdx.x * 128;
  const int wm = (wave & 1) * 64;   // 2 waves along M
  const int wn = (wave >> 1) * 32;  // 4 waves along N

  const int ldr = tid >> 1;         // staging row 0..127
  const int ldk = (tid & 1) * 32;   // staging k-offset 0/32

  v8f acc[4][2];
#pragma unroll
  for (int i = 0; i < 4; i++)
#pragma unroll
    for (int j = 0; j < 2; j++) acc[i][j] = zero8();

  const int aRow = lane & 15;
  const int aK   = (lane >> 4) ? 16 : 0;  // A: one 16-elem run at aK
  const int bCol = lane & 15;
  const int bK   = (lane >> 4) ? 16 : 0;  // B: one 16-elem run at bK

  for (int k0 = 0; k0 < D_; k0 += 64) {
    if (k0 + 64 < D_) {  // prefetch next tiles into L2/WGP$
      __builtin_prefetch(X + (size_t)(m0 + ldr) * D_ + k0 + 64 + ldk, 0, 1);
      __builtin_prefetch(W + (size_t)(n0 + ldr) * D_ + k0 + 64 + ldk, 0, 1);
    }
    stage_cvt32(X + (size_t)(m0 + ldr) * D_ + k0 + ldk, &sA[ldr][ldk]);
    stage_cvt32(W + (size_t)(n0 + ldr) * D_ + k0 + ldk, &sB[ldr][ldk]);
    __syncthreads();

#pragma unroll
    for (int kk = 0; kk < 64; kk += 32) {
      v16bf afrag[4], bfrag[2];
#pragma unroll
      for (int mi = 0; mi < 4; mi++) {
        const __bf16* pa = &sA[wm + mi * 16 + aRow][kk + aK];
        afrag[mi] = frag2(pa, pa + 8);
      }
#pragma unroll
      for (int ni = 0; ni < 2; ni++) {
        const __bf16* pb = &sB[wn + ni * 16 + bCol][kk + bK];
        bfrag[ni] = frag2(pb, pb + 8);
      }
#pragma unroll
      for (int mi = 0; mi < 4; mi++)
#pragma unroll
        for (int ni = 0; ni < 2; ni++)
          acc[mi][ni] = wmma_bf16(afrag[mi], bfrag[ni], acc[mi][ni]);
    }
    __syncthreads();
  }

  // epilogue: add bias, convert, scatter to head-split layout
  const int rOff = (lane >> 4) ? 8 : 0;
#pragma unroll
  for (int mi = 0; mi < 4; mi++) {
#pragma unroll
    for (int ni = 0; ni < 2; ni++) {
      const int n  = n0 + wn + ni * 16 + (lane & 15);
      const float bv = bias[n];
      const int hh = n >> 6;
      const int dk = n & 63;
#pragma unroll
      for (int r = 0; r < 8; r++) {
        const int m = m0 + wm + mi * 16 + r + rOff;
        const int b = m >> 11;
        const int s = m & (S_ - 1);
        const float val = acc[mi][ni][r] + bv;
        size_t addr;
        if (vtrans) addr = ((size_t)(b * H_ + hh) * DK_ + dk) * S_ + s;
        else        addr = ((size_t)(b * H_ + hh) * S_ + s) * DK_ + dk;
        Out[addr] = (__bf16)val;
      }
    }
  }
}

// ---------------------------------------------------------------------------
// Attention: one workgroup = (batch b, 16-row q tile), 16 waves = one per head.
// Loop over all 64 k-steps (32 keys each). Softmax over HEADS via LDS.
// Q/K from [b,h,s,dk] bf16; V from transposed [b,h,dk,s] bf16.
// ---------------------------------------------------------------------------
__global__ __launch_bounds__(512)
void attn_kernel(const __bf16* __restrict__ Qb, const __bf16* __restrict__ Kb,
                 const __bf16* __restrict__ Vt, __bf16* __restrict__ Ab) {
  __shared__ float  sS[16][16][32];  // scores [head][q][k]  (32 KB)
  __shared__ __bf16 sP[16][16][32];  // probs  [head][q][k]  (16 KB)

  const int tid  = threadIdx.x;
  const int lane = tid & 31;
  const int h    = tid >> 5;             // head = wave id
  const int b    = blockIdx.x >> 7;
  const int q0   = (blockIdx.x & 127) * 16;

  const __bf16* Qh = Qb + (size_t)(b * H_ + h) * S_ * DK_;
  const __bf16* Kh = Kb + (size_t)(b * H_ + h) * S_ * DK_;
  const __bf16* Vh = Vt + (size_t)(b * H_ + h) * DK_ * S_;

  const int aRow = lane & 15;
  const int aK   = (lane >> 4) ? 16 : 0;

  // Q A-fragments (16q x 64d = two 16x32 fragments), loaded once
  v16bf qf[2];
#pragma unroll
  for (int dh = 0; dh < 2; dh++) {
    const __bf16* p = Qh + (size_t)(q0 + aRow) * DK_ + dh * 32 + aK;
    qf[dh] = frag2(p, p + 8);
  }

  v8f oacc[4];
#pragma unroll
  for (int i = 0; i < 4; i++) oacc[i] = zero8();

  const int rOff = (lane >> 4) ? 8 : 0;
  const float scale = 0.125f;  // 1/sqrt(DK)

  for (int k0 = 0; k0 < S_; k0 += 32) {
    if (k0 + 32 < S_) {  // prefetch next step's K rows / V row segments
      __builtin_prefetch(Kh + (size_t)(k0 + 32 + lane) * DK_, 0, 1);
      __builtin_prefetch(Vh + (size_t)(lane * 2) * S_ + k0 + 32, 0, 1);
    }

    // --- scores: Q(16x64) @ K^T(64x32) for this head ---
#pragma unroll
    for (int nt = 0; nt < 2; nt++) {
      v8f sc = zero8();
#pragma unroll
      for (int dh = 0; dh < 2; dh++) {
        const int key = k0 + nt * 16 + (lane & 15);
        const __bf16* p =
            Kh + (size_t)key * DK_ + dh * 32 + ((lane >> 4) ? 16 : 0);
        v16bf kf = frag2(p, p + 8);
        sc = wmma_bf16(qf[dh], kf, sc);
      }
#pragma unroll
      for (int r = 0; r < 8; r++)
        sS[h][r + rOff][nt * 16 + (lane & 15)] = sc[r] * scale;
    }
    __syncthreads();

    // --- softmax over heads: 512 threads cover 16q x 32k positions ---
    {
      const int qi = tid >> 5;
      const int ki = tid & 31;
      float vv[16];
      float m = -3.0e38f;
#pragma unroll
      for (int hh = 0; hh < 16; hh++) {
        vv[hh] = sS[hh][qi][ki];
        m = fmaxf(m, vv[hh]);
      }
      float sum = 0.f;
#pragma unroll
      for (int hh = 0; hh < 16; hh++) {
        float e = __expf(vv[hh] - m);
        vv[hh] = e;
        sum += e;
      }
      const float inv = 1.f / sum;
      // masked (k>q): all heads were -1e9 -> softmax over heads = 1/16 each
      const bool masked = (k0 + ki) > (q0 + qi);
#pragma unroll
      for (int hh = 0; hh < 16; hh++)
        sP[hh][qi][ki] = (__bf16)(masked ? 0.0625f : vv[hh] * inv);
    }
    __syncthreads();

    // --- PV: P(16x32) @ V(32x64) ---
    const __bf16* pp = &sP[h][aRow][aK];
    v16bf pf = frag2(pp, pp + 8);
#pragma unroll
    for (int dt = 0; dt < 4; dt++) {
      const int d = dt * 16 + (lane & 15);
      const __bf16* pv = Vh + (size_t)d * S_ + k0 + ((lane >> 4) ? 16 : 0);
      v16bf vf = frag2(pv, pv + 8);
      oacc[dt] = wmma_bf16(pf, vf, oacc[dt]);
    }
  }

  // epilogue: merged-head layout [B,S,D] bf16
#pragma unroll
  for (int dt = 0; dt < 4; dt++) {
#pragma unroll
    for (int r = 0; r < 8; r++) {
      const int q = q0 + r + rOff;
      const int d = dt * 16 + (lane & 15);
      Ab[(size_t)(b * S_ + q) * D_ + h * DK_ + d] = (__bf16)oacc[dt][r];
    }
  }
}

// ---------------------------------------------------------------------------
// Output projection: fp32 Out[4096,1024] = Ab(bf16) @ Wo^T + bo
// A-tile staged via CDNA5 async global->LDS (ASYNCcnt); W converted in-flight.
// ---------------------------------------------------------------------------
__global__ __launch_bounds__(256)
void out_gemm_kernel(const __bf16* __restrict__ A, const float* __restrict__ W,
                     const float* __restrict__ bias, float* __restrict__ Out) {
  __shared__ __bf16 sA[128][64];
  __shared__ __bf16 sB[128][64];

  const int tid  = threadIdx.x;
  const int lane = tid & 31;
  const int wave = tid >> 5;
  const int m0 = blockIdx.y * 128;
  const int n0 = blockIdx.x * 128;
  const int wm = (wave & 1) * 64;
  const int wn = (wave >> 1) * 32;

  const int ldr = tid >> 1;
  const int ldk = (tid & 1) * 32;

  v8f acc[4][2];
#pragma unroll
  for (int i = 0; i < 4; i++)
#pragma unroll
    for (int j = 0; j < 2; j++) acc[i][j] = zero8();

  const int aRow = lane & 15;
  const int aK   = (lane >> 4) ? 16 : 0;
  const int bCol = lane & 15;
  const int bK   = (lane >> 4) ? 16 : 0;

  for (int k0 = 0; k0 < D_; k0 += 64) {
    if (k0 + 64 < D_) {
      __builtin_prefetch(A + (size_t)(m0 + ldr) * D_ + k0 + 64 + ldk, 0, 1);
      __builtin_prefetch(W + (size_t)(n0 + ldr) * D_ + k0 + 64 + ldk, 0, 1);
    }
    { // stage A (already bf16): async global->LDS, 4x16B per thread
      const __bf16* p = A + (size_t)(m0 + ldr) * D_ + k0 + ldk;
      const unsigned l0 = lds_off(&sA[ldr][ldk]);
#pragma unroll
      for (int j = 0; j < 4; j++)
        async_copy_b128(l0 + 16u * j, (const char*)p + 16 * j);
    }
    // stage B = Wo rows, fp32 -> bf16 through VGPRs
    stage_cvt32(W + (size_t)(n0 + ldr) * D_ + k0 + ldk, &sB[ldr][ldk]);
    wait_async0();
    __syncthreads();

#pragma unroll
    for (int kk = 0; kk < 64; kk += 32) {
      v16bf afrag[4], bfrag[2];
#pragma unroll
      for (int mi = 0; mi < 4; mi++) {
        const __bf16* pa = &sA[wm + mi * 16 + aRow][kk + aK];
        afrag[mi] = frag2(pa, pa + 8);
      }
#pragma unroll
      for (int ni = 0; ni < 2; ni++) {
        const __bf16* pb = &sB[wn + ni * 16 + bCol][kk + bK];
        bfrag[ni] = frag2(pb, pb + 8);
      }
#pragma unroll
      for (int mi = 0; mi < 4; mi++)
#pragma unroll
        for (int ni = 0; ni < 2; ni++)
          acc[mi][ni] = wmma_bf16(afrag[mi], bfrag[ni], acc[mi][ni]);
    }
    __syncthreads();
  }

  const int rOff = (lane >> 4) ? 8 : 0;
#pragma unroll
  for (int mi = 0; mi < 4; mi++) {
#pragma unroll
    for (int ni = 0; ni < 2; ni++) {
      const int n = n0 + wn + ni * 16 + (lane & 15);
      const float bv = bias[n];
#pragma unroll
      for (int r = 0; r < 8; r++) {
        const int m = m0 + wm + mi * 16 + r + rOff;
        Out[(size_t)m * D_ + n] = acc[mi][ni][r] + bv;
      }
    }
  }
}

// ---------------------------------------------------------------------------
extern "C" void kernel_launch(void* const* d_in, const int* in_sizes, int n_in,
                              void* d_out, int out_size, void* d_ws,
                              size_t ws_size, hipStream_t stream) {
  const float* q    = (const float*)d_in[0];
  const float* k    = (const float*)d_in[1];
  const float* v    = (const float*)d_in[2];
  // d_in[3] = causal mask; implicit in attn_kernel (k>q -> attn = 1/16)
  const float* wq_w = (const float*)d_in[4];
  const float* wq_b = (const float*)d_in[5];
  const float* wk_w = (const float*)d_in[6];
  const float* wk_b = (const float*)d_in[7];
  const float* wv_w = (const float*)d_in[8];
  const float* wv_b = (const float*)d_in[9];
  const float* wo_w = (const float*)d_in[10];
  const float* wo_b = (const float*)d_in[11];

  const size_t TENS_BYTES = (size_t)B_ * H_ * S_ * DK_ * sizeof(__bf16);  // 8 MB
  char* ws = (char*)d_ws;
  __bf16* Qb = (__bf16*)(ws + 0 * TENS_BYTES);
  __bf16* Kb = (__bf16*)(ws + 1 * TENS_BYTES);
  __bf16* Vt = (__bf16*)(ws + 2 * TENS_BYTES);  // V stored [b,h,dk,s]
  __bf16* Ab = (__bf16*)(ws + 3 * TENS_BYTES);  // attention out [B,S,D]

  dim3 gblk(256);
  dim3 ggrid(D_ / 128, (B_ * S_) / 128);  // 8 x 32

  proj_kernel<<<ggrid, gblk, 0, stream>>>(q, wq_w, wq_b, Qb, 0);
  proj_kernel<<<ggrid, gblk, 0, stream>>>(k, wk_w, wk_b, Kb, 0);
  proj_kernel<<<ggrid, gblk, 0, stream>>>(v, wv_w, wv_b, Vt, 1);

  attn_kernel<<<dim3(B_ * (S_ / 16)), dim3(512), 0, stream>>>(Qb, Kb, Vt, Ab);

  out_gemm_kernel<<<ggrid, gblk, 0, stream>>>(Ab, wo_w, wo_b, (float*)d_out);
}
